// LearnCutV0_58291296141643
// MI455X (gfx1250) — matrix operand order repaired and checked by reading, not compiled, not run on hardware
//
#include <hip/hip_runtime.h>

// ---------------------------------------------------------------------------
// MI455X (gfx1250, wave32) implementation.
// Shapes: B=2, N=2048, M=256, C=256, D=256, DH=128, K=16.
// Strategy: hoist the LN-affine + linear layers out of the (N,M) broadcast so
// the O(N*M) core is a cheap 128-wide VALU kernel; run every dense layer as a
// bf16 WMMA GEMM (v_wmma_f32_16x16x32_bf16, fp32 accumulate). B operands (and
// the transposed-A operand) are pre-packed into bf16 in the exact WMMA lane
// layout so fragment loads are contiguous b128s with no in-loop address math.
// ---------------------------------------------------------------------------

#define Bc 2
#define Nc 2048
#define Mc 256
#define Cc 256
#define Dc 256
#define DH 128
#define Kc 16

typedef __attribute__((ext_vector_type(16))) __bf16 v16bf;
typedef __attribute__((ext_vector_type(8)))  __bf16 v8bf;
typedef __attribute__((ext_vector_type(8)))  float  v8f;

// ---------------------------------------------------------------------------
// Batched WMMA GEMM:  C = act(A @ B + bias)
// A: fp32 (Mdim x KDIM) row-major, or (PACKA) pre-packed bf16 A-layout tiles.
// B: always pre-packed bf16: P[((k/32)*Ndim + n)*32 + (k%32)].
// C: (Mdim x Ndim) fp32 with leading dim ldc.
// Block = 4 waves; each wave owns one 16x32 output tile (2 WMMA / K-step).
// WMMA wave32 layouts per CDNA5 ISA 7.12.2:
//   A frag: lane l -> row m=l&15; elem i -> k=(i>>3)*16+((l>>4)*8)+(i&7)
//   B frag: lane l -> col n=l&15; elem i -> k=(i>>3)*16+((l>>4)*8)+(i&7)  (same as A)
//   C/D:   vgpr r, lane l -> (m = r + (l>>4)*8, n = l&15)
// ---------------------------------------------------------------------------
template <int KDIM, bool PACKA, bool RELU, bool BIAS>
__global__ void gemm_wmma(const void* __restrict__ A,
                          const __bf16* __restrict__ Bp,
                          const float* __restrict__ bias,
                          float* __restrict__ C,
                          int lda, int Ndim, int ldc,
                          long long sA, long long sB, long long sC) {
  const int lane = threadIdx.x & 31;
  const int wave = threadIdx.x >> 5;
  const int tn = blockIdx.x * 32;
  const int tm = (blockIdx.y * 4 + wave) * 16;
  const __bf16* Bb = Bp + (long long)blockIdx.z * sB;
  float* Cb = C + (long long)blockIdx.z * sC;
  const float* Af = (const float*)A + (long long)blockIdx.z * sA;
  const __bf16* Ap = (const __bf16*)A + (long long)blockIdx.z * sA;

  const int am = tm + (lane & 15);   // A row handled by this lane
  const int akb = (lane >> 4) * 8;   // A k sub-offset (ISA layout)
  const int bn0 = tn + (lane & 15);  // first B column
  const int bkb = (lane >> 4) * 8;   // B k sub-offset (ISA layout)

  v8f acc0 = {};
  v8f acc1 = {};
#pragma unroll 4
  for (int k0 = 0; k0 < KDIM; k0 += 32) {
    v16bf afrag;
    if constexpr (PACKA) {
      // packed A: P[(m*(K/32)+kt)*32 + k%32]; lane's halves are contiguous
      const __bf16* ap = Ap + ((size_t)am * (KDIM / 32) + (k0 >> 5)) * 32 + akb;
      v8bf alo = *(const v8bf*)ap;          // k = akb+0..7
      v8bf ahi = *(const v8bf*)(ap + 16);   // k = 16+akb+0..7
      afrag = __builtin_shufflevector(alo, ahi, 0, 1, 2, 3, 4, 5, 6, 7,
                                      8, 9, 10, 11, 12, 13, 14, 15);
    } else {
      // fp32 A: elems 0..7 -> k0+akb+0..7 (contiguous), elems 8..15 -> +16
      const float* arow = Af + (size_t)am * lda + k0 + akb;
      float4 a0 = *(const float4*)(arow + 0);
      float4 a1 = *(const float4*)(arow + 4);
      float4 a2 = *(const float4*)(arow + 16);
      float4 a3 = *(const float4*)(arow + 20);
      afrag[0] = (__bf16)a0.x; afrag[1] = (__bf16)a0.y;
      afrag[2] = (__bf16)a0.z; afrag[3] = (__bf16)a0.w;
      afrag[4] = (__bf16)a1.x; afrag[5] = (__bf16)a1.y;
      afrag[6] = (__bf16)a1.z; afrag[7] = (__bf16)a1.w;
      afrag[8] = (__bf16)a2.x; afrag[9] = (__bf16)a2.y;
      afrag[10] = (__bf16)a2.z; afrag[11] = (__bf16)a2.w;
      afrag[12] = (__bf16)a3.x; afrag[13] = (__bf16)a3.y;
      afrag[14] = (__bf16)a3.z; afrag[15] = (__bf16)a3.w;
    }
    // packed B: lane's halves k = bkb+0..7 and 16+bkb+0..7 (two 16-byte loads)
    const __bf16* bp = Bb + ((size_t)(k0 >> 5) * Ndim + bn0) * 32 + bkb;
    v16bf bf0 = __builtin_shufflevector(*(const v8bf*)bp, *(const v8bf*)(bp + 16),
                                        0, 1, 2, 3, 4, 5, 6, 7, 8, 9, 10, 11, 12, 13, 14, 15);
    v16bf bf1 = __builtin_shufflevector(*(const v8bf*)(bp + 16 * 32), *(const v8bf*)(bp + 16 * 32 + 16),
                                        0, 1, 2, 3, 4, 5, 6, 7, 8, 9, 10, 11, 12, 13, 14, 15);  // column bn0+16
    acc0 = __builtin_amdgcn_wmma_f32_16x16x32_bf16(
        false, afrag, false, bf0, (short)0, acc0, false, false);
    acc1 = __builtin_amdgcn_wmma_f32_16x16x32_bf16(
        false, afrag, false, bf1, (short)0, acc1, false, false);
  }

  float bv0 = 0.f, bv1 = 0.f;
  if constexpr (BIAS) { bv0 = bias[bn0]; bv1 = bias[bn0 + 16]; }
#pragma unroll
  for (int r = 0; r < 8; ++r) {
    int m = tm + r + ((lane >> 4) * 8);
    float v0 = acc0[r];
    float v1 = acc1[r];
    if constexpr (BIAS) { v0 += bv0; v1 += bv1; }
    if constexpr (RELU) { v0 = fmaxf(v0, 0.f); v1 = fmaxf(v1, 0.f); }
    Cb[(size_t)m * ldc + bn0] = v0;
    Cb[(size_t)m * ldc + bn0 + 16] = v1;
  }
}

template <int KDIM, bool PACKA, bool RELU, bool BIAS>
static void launch_gemm(const void* A, const __bf16* Bp, const float* bias, float* C,
                        int Mdim, int Ndim, int lda, int ldc, int nbatch,
                        long long sA, long long sB, long long sC, hipStream_t stream) {
  dim3 grid(Ndim / 32, Mdim / 64, nbatch);
  gemm_wmma<KDIM, PACKA, RELU, BIAS>
      <<<grid, dim3(128), 0, stream>>>(A, Bp, bias, C, lda, Ndim, ldc, sA, sB, sC);
}

// float (K x N) row-major -> bf16 packed P[((k/32)*N + n)*32 + k%32], batched
__global__ void pack_b_kernel(const float* __restrict__ B, __bf16* __restrict__ P,
                              int K, int N, int ldb, long long sB, long long sP) {
  long long idx = (long long)blockIdx.x * blockDim.x + threadIdx.x;
  if (idx >= (long long)K * N) return;
  int b = blockIdx.y;
  int kk = (int)(idx & 31);
  long long t = idx >> 5;
  int n = (int)(t % N);
  int kt = (int)(t / N);
  P[(long long)b * sP + idx] = (__bf16)B[(long long)b * sB + (size_t)(kt * 32 + kk) * ldb + n];
}

// pack G^T (per batch) into A-layout: P[(m*(K/32)+kt)*32 + kk] = G[(kt*32+kk)*Mc + m]
__global__ void pack_at_kernel(const float* __restrict__ G, __bf16* __restrict__ P,
                               long long sG, long long sP) {
  long long idx = (long long)blockIdx.x * blockDim.x + threadIdx.x;
  if (idx >= (long long)Mc * Nc) return;
  int b = blockIdx.y;
  int kk = (int)(idx & 31);
  long long t = idx >> 5;
  int kt = (int)(t % (Nc / 32));
  int m = (int)(t / (Nc / 32));
  P[(long long)b * sP + idx] = (__bf16)G[(long long)b * sG + (size_t)(kt * 32 + kk) * Mc + m];
}

// ---------------------------------------------------------------------------
// rows x DH = xyz(rows,3) @ W(3,DH) (+ bias)
// ---------------------------------------------------------------------------
__global__ void linear3_kernel(const float* __restrict__ xyz,
                               const float* __restrict__ W,
                               const float* __restrict__ bias,
                               float* __restrict__ out, int rows) {
  int idx = blockIdx.x * blockDim.x + threadIdx.x;
  if (idx >= rows * DH) return;
  int r = idx >> 7, c = idx & (DH - 1);
  float v = bias ? bias[c] : 0.f;
  v += xyz[r * 3 + 0] * W[c] + xyz[r * 3 + 1] * W[DH + c] + xyz[r * 3 + 2] * W[2 * DH + c];
  out[idx] = v;
}

// W2p[c,c'] = ln_g[c] * w2[c,c']
__global__ void build_w2p_kernel(const float* __restrict__ w2,
                                 const float* __restrict__ g,
                                 float* __restrict__ w2p) {
  int idx = blockIdx.x * blockDim.x + threadIdx.x;
  if (idx >= DH * DH) return;
  w2p[idx] = g[idx >> 7] * w2[idx];
}

// s[c'] = sum_c g[c]*w2[c,c'] ; q[c'] = sum_c ln_b[c]*w2[c,c'] + b2[c']
__global__ void build_sq_kernel(const float* __restrict__ w2,
                                const float* __restrict__ g,
                                const float* __restrict__ beta,
                                const float* __restrict__ b2,
                                float* __restrict__ svec, float* __restrict__ qvec) {
  int cp = threadIdx.x;
  float s = 0.f, q = 0.f;
  for (int c = 0; c < DH; ++c) {
    float w = w2[c * DH + cp];
    s += g[c] * w;
    q += beta[c] * w;
  }
  svec[cp] = s;
  qvec[cp] = q + b2[cp];
}

// ---------------------------------------------------------------------------
// Row LayerNorm (width 128 or 256), optional relu after LN.
// ---------------------------------------------------------------------------
__global__ void ln_rows_kernel(const float* __restrict__ X,
                               const float* __restrict__ gain,
                               const float* __restrict__ beta,
                               float* __restrict__ Y, int width, int reluAfter) {
  __shared__ float r1[256];
  __shared__ float r2[256];
  int row = blockIdx.x, t = threadIdx.x;
  float x = X[(size_t)row * width + t];
  r1[t] = x;
  r2[t] = x * x;
  __syncthreads();
  for (int s = width >> 1; s > 0; s >>= 1) {
    if (t < s) { r1[t] += r1[t + s]; r2[t] += r2[t + s]; }
    __syncthreads();
  }
  float inv = 1.f / (float)width;
  float mu = r1[0] * inv;
  float var = r2[0] * inv - mu * mu;
  float y = (x - mu) * rsqrtf(var + 1e-5f) * gain[t] + beta[t];
  if (reluAfter) y = fmaxf(y, 0.f);
  Y[(size_t)row * width + t] = y;
}

__device__ inline float wave_sum32(float v) {
#pragma unroll
  for (int off = 16; off > 0; off >>= 1) v += __shfl_xor(v, off, 32);
  return v;
}

// ---------------------------------------------------------------------------
// Pairwise logits: one wave per point n, loops over all M superpoints.
// logit[n,m] = <relu((u_n - v_m - mu*s)*inv_sigma + q), g_pi_m>
//            * <relu(pv_n - sv_m), h_fi_m>
// mu/sigma are the LN stats of (a_n - cc_m), computed by butterfly reduction.
// ---------------------------------------------------------------------------
__global__ void pair_logits_kernel(const float* __restrict__ a,
                                   const float* __restrict__ cc,
                                   const float* __restrict__ u,
                                   const float* __restrict__ v,
                                   const float* __restrict__ pv,
                                   const float* __restrict__ sv,
                                   const float* __restrict__ gpi,
                                   const float* __restrict__ hfi,
                                   const float* __restrict__ svec,
                                   const float* __restrict__ qvec,
                                   float* __restrict__ logits) {
  const int lane = threadIdx.x & 31;
  const int n = blockIdx.x * 8 + (threadIdx.x >> 5);
  const int b = blockIdx.y;
  const size_t prow = (size_t)(b * Nc + n);

  float4 an  = ((const float4*)(a  + prow * DH))[lane];
  float4 un  = ((const float4*)(u  + prow * DH))[lane];
  float4 pvn = ((const float4*)(pv + prow * DH))[lane];
  float4 s4  = ((const float4*)svec)[lane];
  float4 q4  = ((const float4*)qvec)[lane];

  for (int m = 0; m < Mc; ++m) {
    const size_t srow = (size_t)(b * Mc + m);
    float4 cm = ((const float4*)(cc  + srow * DH))[lane];
    float4 vm = ((const float4*)(v   + srow * DH))[lane];
    float4 sm = ((const float4*)(sv  + srow * DH))[lane];
    float4 gm = ((const float4*)(gpi + srow * DH))[lane];
    float4 hm = ((const float4*)(hfi + srow * DH))[lane];

    float x0 = an.x - cm.x, x1 = an.y - cm.y, x2 = an.z - cm.z, x3 = an.w - cm.w;
    float sx  = wave_sum32(x0 + x1 + x2 + x3);
    float sx2 = wave_sum32(x0 * x0 + x1 * x1 + x2 * x2 + x3 * x3);
    float mu = sx * (1.f / DH);
    float var = sx2 * (1.f / DH) - mu * mu;
    float is = rsqrtf(var + 1e-5f);

    float p = fmaxf((un.x - vm.x - mu * s4.x) * is + q4.x, 0.f) * gm.x
            + fmaxf((un.y - vm.y - mu * s4.y) * is + q4.y, 0.f) * gm.y
            + fmaxf((un.z - vm.z - mu * s4.z) * is + q4.z, 0.f) * gm.z
            + fmaxf((un.w - vm.w - mu * s4.w) * is + q4.w, 0.f) * gm.w;
    float w = fmaxf(pvn.x - sm.x, 0.f) * hm.x
            + fmaxf(pvn.y - sm.y, 0.f) * hm.y
            + fmaxf(pvn.z - sm.z, 0.f) * hm.z
            + fmaxf(pvn.w - sm.w, 0.f) * hm.w;
    float pd = wave_sum32(p);
    float vd = wave_sum32(w);
    if (lane == 0) logits[prow * Mc + m] = pd * vd;
  }
}

// ---------------------------------------------------------------------------
// Per point n: squared distances, top-16 rank mask (top_k tie-break by index),
// softmax over all M, mask, renormalize by (sum + 1e-4). Block = 256 threads.
// ---------------------------------------------------------------------------
__global__ void softmax_mask_kernel(const float* __restrict__ p_xyz,
                                    const float* __restrict__ sp_xyz,
                                    const float* __restrict__ logits,
                                    float* __restrict__ G) {
  __shared__ float dist[Mc];
  __shared__ float red[Mc];
  const int b = blockIdx.y, n = blockIdx.x, m = threadIdx.x;
  const size_t prow = (size_t)(b * Nc + n);

  float px = p_xyz[prow * 3 + 0], py = p_xyz[prow * 3 + 1], pz = p_xyz[prow * 3 + 2];
  float dx = px - sp_xyz[(size_t)(b * Mc + m) * 3 + 0];
  float dy = py - sp_xyz[(size_t)(b * Mc + m) * 3 + 1];
  float dz = pz - sp_xyz[(size_t)(b * Mc + m) * 3 + 2];
  float d2 = dx * dx + dy * dy + dz * dz;
  dist[m] = d2;
  __syncthreads();
  int rank = 0;
  for (int j = 0; j < Mc; ++j) {
    float dj = dist[j];
    rank += (dj < d2) || (dj == d2 && j < m);
  }
  bool keep = rank < Kc;

  float w = logits[prow * Mc + m];
  red[m] = w;
  __syncthreads();
  for (int s = Mc >> 1; s > 0; s >>= 1) {
    if (m < s) red[m] = fmaxf(red[m], red[m + s]);
    __syncthreads();
  }
  float wmax = red[0];
  __syncthreads();
  float e = __expf(w - wmax);
  red[m] = e;
  __syncthreads();
  for (int s = Mc >> 1; s > 0; s >>= 1) {
    if (m < s) red[m] += red[m + s];
    __syncthreads();
  }
  float esum = red[0];
  __syncthreads();
  float gh = keep ? (e / esum) : 0.f;
  red[m] = gh;
  __syncthreads();
  for (int s = Mc >> 1; s > 0; s >>= 1) {
    if (m < s) red[m] += red[m + s];
    __syncthreads();
  }
  float gsum = red[0];
  G[prow * Mc + m] = gh / (gsum + 1e-4f);
}

// Gcol[b,m] = sum_n G[b,n,m]  (coalesced, deterministic)
__global__ void colsum_kernel(const float* __restrict__ G, float* __restrict__ Gcol) {
  int b = blockIdx.y, m = threadIdx.x;
  const float* Gb = G + (size_t)b * Nc * Mc;
  float s = 0.f;
  for (int n = 0; n < Nc; ++n) s += Gb[(size_t)n * Mc + m];
  Gcol[b * Mc + m] = s;
}

// cat[b,m,:C]=sp_fea, cat[b,m,C:]=spp/(Gcol+1e-8)
__global__ void norm_concat_kernel(const float* __restrict__ sp_fea,
                                   const float* __restrict__ spp,
                                   const float* __restrict__ Gcol,
                                   float* __restrict__ cat) {
  int idx = blockIdx.x * blockDim.x + threadIdx.x;
  if (idx >= Bc * Mc * Cc) return;
  int c = idx & (Cc - 1);
  int row = idx >> 8;  // b*Mc + m
  float denom = Gcol[row] + 1e-8f;
  cat[(size_t)row * (2 * Cc) + c] = sp_fea[idx];
  cat[(size_t)row * (2 * Cc) + Cc + c] = spp[idx] / denom;
}

// out[b,n,256:512] = p_fea
__global__ void copy_pfea_kernel(const float* __restrict__ p_fea, float* __restrict__ out) {
  int idx = blockIdx.x * blockDim.x + threadIdx.x;
  if (idx >= Bc * Nc * Cc) return;
  int c = idx & (Cc - 1);
  int row = idx >> 8;  // b*Nc + n
  out[(size_t)row * (Dc + Cc) + Dc + c] = p_fea[idx];
}

// ---------------------------------------------------------------------------
extern "C" void kernel_launch(void* const* d_in, const int* in_sizes, int n_in,
                              void* d_out, int out_size, void* d_ws, size_t ws_size,
                              hipStream_t stream) {
  (void)in_sizes; (void)n_in; (void)out_size; (void)ws_size;
  const float* p_xyz   = (const float*)d_in[0];
  const float* p_fea   = (const float*)d_in[1];
  const float* sp_xyz  = (const float*)d_in[2];
  const float* sp_fea  = (const float*)d_in[3];
  const float* phi_w1  = (const float*)d_in[4];
  const float* phi_b1  = (const float*)d_in[5];
  const float* phi_lng = (const float*)d_in[6];
  const float* phi_lnb = (const float*)d_in[7];
  const float* phi_w2  = (const float*)d_in[8];
  const float* phi_b2  = (const float*)d_in[9];
  const float* varphi_w = (const float*)d_in[10];
  const float* varphi_b = (const float*)d_in[11];
  const float* g_w1 = (const float*)d_in[12];
  const float* g_b1 = (const float*)d_in[13];
  const float* g_lng = (const float*)d_in[14];
  const float* g_lnb = (const float*)d_in[15];
  const float* g_w2 = (const float*)d_in[16];
  const float* g_b2 = (const float*)d_in[17];
  const float* h_w1 = (const float*)d_in[18];
  const float* h_b1 = (const float*)d_in[19];
  const float* h_lng = (const float*)d_in[20];
  const float* h_lnb = (const float*)d_in[21];
  const float* h_w2 = (const float*)d_in[22];
  const float* h_b2 = (const float*)d_in[23];
  const float* fus_w1 = (const float*)d_in[24];
  const float* fus_b1 = (const float*)d_in[25];
  const float* fus_lng = (const float*)d_in[26];
  const float* fus_lnb = (const float*)d_in[27];
  const float* fus_w2 = (const float*)d_in[28];
  const float* fus_b2 = (const float*)d_in[29];
  float* out = (float*)d_out;

  // ---- workspace carve: fp32 pool ----
  float* ws = (float*)d_ws;
  size_t off = 0;
  float* a_buf = ws + off; off += (size_t)Bc * Nc * DH;
  float* u_buf = ws + off; off += (size_t)Bc * Nc * DH;
  float* pv_buf = ws + off; off += (size_t)Bc * Nc * DH;
  float* cc_buf = ws + off; off += (size_t)Bc * Mc * DH;
  float* v_buf = ws + off; off += (size_t)Bc * Mc * DH;
  float* sv_buf = ws + off; off += (size_t)Bc * Mc * DH;
  float* g1_buf = ws + off; off += (size_t)Bc * Mc * DH;
  float* gln_buf = ws + off; off += (size_t)Bc * Mc * DH;
  float* gpi_buf = ws + off; off += (size_t)Bc * Mc * DH;
  float* h1_buf = ws + off; off += (size_t)Bc * Mc * DH;
  float* hln_buf = ws + off; off += (size_t)Bc * Mc * DH;
  float* hfi_buf = ws + off; off += (size_t)Bc * Mc * DH;
  float* w2p_buf = ws + off; off += (size_t)DH * DH;
  float* s_buf = ws + off; off += DH;
  float* q_buf = ws + off; off += DH;
  float* logit_buf = ws + off; off += (size_t)Bc * Nc * Mc;
  float* G_buf = ws + off; off += (size_t)Bc * Nc * Mc;
  float* Gcol_buf = ws + off; off += (size_t)Bc * Mc;
  float* spp_buf = ws + off; off += (size_t)Bc * Mc * Cc;
  float* cat_buf = ws + off; off += (size_t)Bc * Mc * 2 * Cc;
  float* f_buf = ws + off; off += (size_t)Bc * Mc * Dc;
  float* fln_buf = ws + off; off += (size_t)Bc * Mc * Dc;
  float* spnew_buf = ws + off; off += (size_t)Bc * Mc * Dc;
  // ---- bf16 pool (32-byte aligned) ----
  off = (off + 7) & ~(size_t)7;
  __bf16* bws = (__bf16*)(ws + off);
  size_t boff = 0;
  __bf16* pb_w2p = bws + boff; boff += (size_t)DH * DH;
  __bf16* pb_varphi = bws + boff; boff += (size_t)Cc * DH;
  __bf16* pb_gw2 = bws + boff; boff += (size_t)DH * DH;
  __bf16* pb_hw1 = bws + boff; boff += (size_t)Cc * DH;
  __bf16* pb_hw2 = bws + boff; boff += (size_t)DH * DH;
  __bf16* pb_fw1 = bws + boff; boff += (size_t)2 * Cc * Dc;
  __bf16* pb_fw2 = bws + boff; boff += (size_t)Dc * Dc;
  __bf16* pb_pfea = bws + boff; boff += (size_t)Bc * Nc * Cc;
  __bf16* pb_spnew = bws + boff; boff += (size_t)Bc * Mc * Dc;
  __bf16* pa_G = bws + boff; boff += (size_t)Bc * Mc * Nc;

  // --- precompute: W2-hoist of phi's LayerNorm ---
  build_w2p_kernel<<<(DH * DH + 255) / 256, 256, 0, stream>>>(phi_w2, phi_lng, w2p_buf);
  build_sq_kernel<<<1, DH, 0, stream>>>(phi_w2, phi_lng, phi_lnb, phi_b2, s_buf, q_buf);

  // --- pack static B operands (weights) to bf16 WMMA layout ---
  pack_b_kernel<<<dim3((DH * DH + 255) / 256, 1), 256, 0, stream>>>(
      w2p_buf, pb_w2p, DH, DH, DH, 0, 0);
  pack_b_kernel<<<dim3((Cc * DH + 255) / 256, 1), 256, 0, stream>>>(
      varphi_w, pb_varphi, Cc, DH, DH, 0, 0);
  pack_b_kernel<<<dim3((DH * DH + 255) / 256, 1), 256, 0, stream>>>(
      g_w2, pb_gw2, DH, DH, DH, 0, 0);
  pack_b_kernel<<<dim3((Cc * DH + 255) / 256, 1), 256, 0, stream>>>(
      h_w1, pb_hw1, Cc, DH, DH, 0, 0);
  pack_b_kernel<<<dim3((DH * DH + 255) / 256, 1), 256, 0, stream>>>(
      h_w2, pb_hw2, DH, DH, DH, 0, 0);
  pack_b_kernel<<<dim3((2 * Cc * Dc + 255) / 256, 1), 256, 0, stream>>>(
      fus_w1, pb_fw1, 2 * Cc, Dc, Dc, 0, 0);
  pack_b_kernel<<<dim3((Dc * Dc + 255) / 256, 1), 256, 0, stream>>>(
      fus_w2, pb_fw2, Dc, Dc, Dc, 0, 0);
  pack_b_kernel<<<dim3((Nc * Cc + 255) / 256, Bc), 256, 0, stream>>>(
      p_fea, pb_pfea, Nc, Cc, Cc, (long long)Nc * Cc, (long long)Nc * Cc);

  // --- tiny 3->128 linears ---
  linear3_kernel<<<(Bc * Nc * DH + 255) / 256, 256, 0, stream>>>(p_xyz, phi_w1, phi_b1, a_buf, Bc * Nc);
  linear3_kernel<<<(Bc * Mc * DH + 255) / 256, 256, 0, stream>>>(sp_xyz, phi_w1, nullptr, cc_buf, Bc * Mc);
  linear3_kernel<<<(Bc * Mc * DH + 255) / 256, 256, 0, stream>>>(sp_xyz, g_w1, g_b1, g1_buf, Bc * Mc);
  ln_rows_kernel<<<Bc * Mc, DH, 0, stream>>>(g1_buf, g_lng, g_lnb, gln_buf, DH, 1);

  // --- WMMA GEMMs ---
  // u = a @ W2'            (4096 x 128, K=128)
  launch_gemm<DH, false, false, false>(a_buf, pb_w2p, nullptr, u_buf,
      Bc * Nc, DH, DH, DH, 1, 0, 0, 0, stream);
  // v = cc @ W2'           (512 x 128, K=128)
  launch_gemm<DH, false, false, false>(cc_buf, pb_w2p, nullptr, v_buf,
      Bc * Mc, DH, DH, DH, 1, 0, 0, 0, stream);
  // pv = p_fea @ varphi_w + b   (4096 x 128, K=256)
  launch_gemm<Cc, false, false, true>(p_fea, pb_varphi, varphi_b, pv_buf,
      Bc * Nc, DH, Cc, DH, 1, 0, 0, 0, stream);
  // sv = sp_fea @ varphi_w      (512 x 128, K=256)
  launch_gemm<Cc, false, false, false>(sp_fea, pb_varphi, nullptr, sv_buf,
      Bc * Mc, DH, Cc, DH, 1, 0, 0, 0, stream);
  // g_pi = relu(LN(g1)) @ g_w2 + g_b2   (512 x 128, K=128)
  launch_gemm<DH, false, false, true>(gln_buf, pb_gw2, g_b2, gpi_buf,
      Bc * Mc, DH, DH, DH, 1, 0, 0, 0, stream);
  // h1 = relu(sp_fea @ h_w1 + h_b1)     (512 x 128, K=256)
  launch_gemm<Cc, false, true, true>(sp_fea, pb_hw1, h_b1, h1_buf,
      Bc * Mc, DH, Cc, DH, 1, 0, 0, 0, stream);
  ln_rows_kernel<<<Bc * Mc, DH, 0, stream>>>(h1_buf, h_lng, h_lnb, hln_buf, DH, 0);
  // h_fi = LN(h1) @ h_w2 + h_b2         (512 x 128, K=128)
  launch_gemm<DH, false, false, true>(hln_buf, pb_hw2, h_b2, hfi_buf,
      Bc * Mc, DH, DH, DH, 1, 0, 0, 0, stream);

  // --- pairwise logits (wave per point) and softmax/top-k ---
  pair_logits_kernel<<<dim3(Nc / 8, Bc), 256, 0, stream>>>(
      a_buf, cc_buf, u_buf, v_buf, pv_buf, sv_buf, gpi_buf, hfi_buf, s_buf, q_buf, logit_buf);
  softmax_mask_kernel<<<dim3(Nc, Bc), Mc, 0, stream>>>(p_xyz, sp_xyz, logit_buf, G_buf);
  colsum_kernel<<<dim3(1, Bc), Mc, 0, stream>>>(G_buf, Gcol_buf);

  // --- sp_fea_plus = (G^T @ p_fea) / Gsum, fusion MLP ---
  pack_at_kernel<<<dim3((Mc * Nc + 255) / 256, Bc), 256, 0, stream>>>(
      G_buf, pa_G, (long long)Nc * Mc, (long long)Mc * Nc);
  // spp[b] = packed(G[b]^T) (256 x 2048) @ packed(p_fea[b]) -> (256 x 256)
  launch_gemm<Nc, true, false, false>(pa_G, pb_pfea, nullptr, spp_buf,
      Mc, Cc, 0, Cc, Bc,
      (long long)Mc * Nc, (long long)Nc * Cc, (long long)Mc * Cc, stream);
  norm_concat_kernel<<<(Bc * Mc * Cc + 255) / 256, 256, 0, stream>>>(
      sp_fea, spp_buf, Gcol_buf, cat_buf);
  // f = relu(cat @ fus_w1 + fus_b1)   (512 x 256, K=512)
  launch_gemm<2 * Cc, false, true, true>(cat_buf, pb_fw1, fus_b1, f_buf,
      Bc * Mc, Dc, 2 * Cc, Dc, 1, 0, 0, 0, stream);
  ln_rows_kernel<<<Bc * Mc, Dc, 0, stream>>>(f_buf, fus_lng, fus_lnb, fln_buf, Dc, 0);
  // sp_fea_new = LN(f) @ fus_w2 + fus_b2   (512 x 256, K=256)
  launch_gemm<Dc, false, false, true>(fln_buf, pb_fw2, fus_b2, spnew_buf,
      Bc * Mc, Dc, Dc, Dc, 1, 0, 0, 0, stream);
  pack_b_kernel<<<dim3((Mc * Dc + 255) / 256, Bc), 256, 0, stream>>>(
      spnew_buf, pb_spnew, Mc, Dc, Dc, (long long)Mc * Dc, (long long)Mc * Dc);

  // --- l2g = G @ sp_fea_new -> out[:, :, 0:256]; p_fea -> out[:, :, 256:512] ---
  launch_gemm<Mc, false, false, false>(G_buf, pb_spnew, nullptr, out,
      Nc, Dc, Mc, Dc + Cc, Bc,
      (long long)Nc * Mc, (long long)Mc * Dc, (long long)Nc * (Dc + Cc), stream);
  copy_pfea_kernel<<<(Bc * Nc * Cc + 255) / 256, 256, 0, stream>>>(p_fea, out);
}
